// RNN_32435593019448
// MI455X (gfx1250) — matrix operand order repaired and checked by reading, not compiled, not run on hardware
//
#include <hip/hip_runtime.h>

// ---------------------------------------------------------------- constants
#define GB 64     // batch
#define GS 512    // seq len
#define GD 512    // input dim
#define GH 512    // hidden dim

typedef __attribute__((ext_vector_type(16))) __bf16 v16bf;
typedef __attribute__((ext_vector_type(8)))  float  v8f;
typedef __attribute__((ext_vector_type(2)))  float  f32x2;
typedef __attribute__((ext_vector_type(2)))  __bf16 bf16x2;

#define WMMA_BF16(a, b, c) \
  __builtin_amdgcn_wmma_f32_16x16x32_bf16(false, (a), false, (b), (short)0, (c), false, false)

// packed f32x2 -> bf16x2 (native v_cvt_pk_bf16_f32 on gfx1250)
__device__ __forceinline__ unsigned pack_bf2(float a, float b) {
  union { bf16x2 v; unsigned u; } r;
  r.v = __builtin_convertvector((f32x2){a, b}, bf16x2);
  return r.u;
}
__device__ __forceinline__ unsigned short bf1(float a) {
  union { __bf16 b; unsigned short u; } r;
  r.b = (__bf16)a;
  return r.u;
}

// A fragment (16x32 bf16, MxK): lanes 0-15 -> M=lane, interleaved K pattern.
__device__ __forceinline__ v16bf frag_a(const unsigned short* p0, int row, int kbase, int stride) {
  union { v16bf v; unsigned u[8]; } r;
  const int lane = threadIdx.x & 31;
  const int half = lane >> 4;
  const unsigned short* p = p0 + (row + (lane & 15)) * stride + kbase;
#pragma unroll
  for (int v = 0; v < 8; ++v) {
    const int k = ((v >> 2) << 4) + (half << 3) + ((v & 3) << 1);
    r.u[v] = *(const unsigned*)(p + k);
  }
  return r.v;
}

// B fragment (32x16 bf16, KxN): same interleaved K pattern as frag_a.
// LDS layout is [n][k] so K pairs are contiguous.
__device__ __forceinline__ v16bf frag_b(const unsigned short* p0, int col, int kbase, int stride) {
  union { v16bf v; unsigned u[8]; } r;
  const int lane = threadIdx.x & 31;
  const int half = lane >> 4;
  const unsigned short* p = p0 + (col + (lane & 15)) * stride + kbase + (half << 3);
#pragma unroll
  for (int v = 0; v < 8; ++v) r.u[v] = *(const unsigned*)(p + ((v >> 2) << 4) + ((v & 3) << 1));
  return r.v;
}

// ---------------------------------------------------------------- gate GEMM
// out_g[m,n] = sum_k X[m,k] * W_g[k,n] + bias_g[n];  M=32768, N=512, K=512
// Block tile 128x128, 256 threads (8 waves, 2x4), wave tile 64x32, BK=32.
// Software-pipelined: global loads of tile k+1 overlap WMMA of tile k.
__global__ __launch_bounds__(256) void gemm_gates(
    const float* __restrict__ X,
    const float* __restrict__ W0, const float* __restrict__ W1, const float* __restrict__ W2,
    const float* __restrict__ b0, const float* __restrict__ b1, const float* __restrict__ b2,
    float* __restrict__ o0, float* __restrict__ o1, float* __restrict__ o2) {
  const int g = blockIdx.z;
  const float* W    = (g == 0) ? W0 : ((g == 1) ? W1 : W2);
  const float* bias = (g == 0) ? b0 : ((g == 1) ? b1 : b2);
  float*       out  = (g == 0) ? o0 : ((g == 1) ? o1 : o2);

  __shared__ __align__(16) unsigned short As[128 * 40];  // [m][k], stride 40
  __shared__ __align__(16) unsigned short Bs[128 * 40];  // [n][k], stride 40

  const int tid   = threadIdx.x;
  const int wid   = tid >> 5;
  const int waveM = wid >> 2;   // 0..1 -> 64 rows
  const int waveN = wid & 3;    // 0..3 -> 32 cols
  const int m0 = blockIdx.y * 128;
  const int n0 = blockIdx.x * 128;

  v8f acc[4][2];
#pragma unroll
  for (int i = 0; i < 4; ++i)
#pragma unroll
    for (int j = 0; j < 2; ++j) acc[i][j] = (v8f){0.f, 0.f, 0.f, 0.f, 0.f, 0.f, 0.f, 0.f};

  const int bn  = tid & 127;   // B-tile: column owned by this thread
  const int bkh = tid >> 7;    // 0..1 -> k half (16 rows each)

  float4 ar[4];                // raw A staging (tile k+1)
  float  brg[16];              // raw B staging (tile k+1)

  auto loadTile = [&](int kt) {
    const int k0 = kt * 32;
#pragma unroll
    for (int i = 0; i < 4; ++i) {
      const int slot = tid + i * 256;
      const int row = slot >> 3, c4 = slot & 7;
      ar[i] = *(const float4*)(X + (size_t)(m0 + row) * GD + k0 + c4 * 4);
    }
#pragma unroll
    for (int j = 0; j < 16; ++j)
      brg[j] = W[(size_t)(k0 + bkh * 16 + j) * GH + n0 + bn];
  };

  auto storeTile = [&]() {
#pragma unroll
    for (int i = 0; i < 4; ++i) {
      const int slot = tid + i * 256;
      const int row = slot >> 3, c4 = slot & 7;
      unsigned* d = (unsigned*)(As + row * 40 + c4 * 4);
      d[0] = pack_bf2(ar[i].x, ar[i].y);
      d[1] = pack_bf2(ar[i].z, ar[i].w);
    }
    unsigned* d = (unsigned*)(Bs + bn * 40 + bkh * 16);
#pragma unroll
    for (int j = 0; j < 8; ++j) d[j] = pack_bf2(brg[2 * j], brg[2 * j + 1]);
  };

  loadTile(0);
#pragma unroll 1
  for (int kt = 0; kt < 16; ++kt) {
    __syncthreads();           // previous compute finished reading LDS
    storeTile();               // convert + store tile kt
    __syncthreads();
    if (kt < 15) loadTile(kt + 1);   // issue global loads; waits land after WMMAs

    v16bf bF[2];
#pragma unroll
    for (int nt = 0; nt < 2; ++nt) bF[nt] = frag_b(Bs, waveN * 32 + nt * 16, 0, 40);
#pragma unroll
    for (int mt = 0; mt < 4; ++mt) {
      const v16bf aF = frag_a(As, waveM * 64 + mt * 16, 0, 40);
#pragma unroll
      for (int nt = 0; nt < 2; ++nt) acc[mt][nt] = WMMA_BF16(aF, bF[nt], acc[mt][nt]);
    }
  }

  // epilogue: C layout M = v + 8*half, N = lane%16
  const int lane = tid & 31, half = lane >> 4, nl = lane & 15;
#pragma unroll
  for (int mt = 0; mt < 4; ++mt)
#pragma unroll
    for (int nt = 0; nt < 2; ++nt) {
      const int gn = n0 + waveN * 32 + nt * 16 + nl;
      const float bs = bias[gn];
#pragma unroll
      for (int v = 0; v < 8; ++v) {
        const int gm = m0 + waveM * 64 + mt * 16 + v + 8 * half;
        out[(size_t)gm * GH + gn] = acc[mt][nt][v] + bs;
      }
    }
}

// ---------------------------------------------------------------- recurrent scan
// Persistent kernel: 32 blocks x 128 threads. Each block owns 16 hidden cols,
// each of 4 waves owns a 16-batch tile. The f32 hidden state is carried in
// REGISTERS (each (b,n) element is lane-private across all 512 steps); only the
// bf16 image of h is exchanged through global memory (ping-pong, 64 KB) and
// copied to LDS each step with zero-VALU b128 copies. Recurrent weights
// (3 x 512 x 16, bf16) live in LDS for all steps. Dynamic LDS = 116480 B.
__global__ __launch_bounds__(128) void gru_scan(
    const float* __restrict__ gxr, const float* __restrict__ gxz, const float* __restrict__ gxh,
    const float* __restrict__ Wr, const float* __restrict__ Wz, const float* __restrict__ Wh,
    const float* __restrict__ h_init,              // (B, L, H), pre-offset to layer
    unsigned short* __restrict__ hbfA, unsigned short* __restrict__ hbfB,
    float* __restrict__ ys, unsigned* __restrict__ sync) {
  extern __shared__ __align__(16) unsigned short smem[];
  unsigned short* Wl = smem;                   // [g][n][k], 3*16*520
  unsigned short* Hs = smem + 3 * 16 * 520;    // [b][k],    64*520 (offset 49920 B, 16-aligned)

  const int tid  = threadIdx.x;
  const int wid  = tid >> 5;
  const int lane = tid & 31;
  const int half = lane >> 4;
  const int nl   = lane & 15;
  const int n0   = blockIdx.x * 16;
  const int gn   = n0 + nl;

  // one-time: recurrent weight slice -> LDS bf16
  {
    const int n  = tid & 15;
    const int kq = tid >> 4;  // 0..7
    const float* Wg[3] = {Wr, Wz, Wh};
#pragma unroll
    for (int g = 0; g < 3; ++g)
#pragma unroll 4
      for (int i = 0; i < 64; ++i) {
        const int k = kq * 64 + i;
        Wl[(g * 16 + n) * 520 + k] = bf1(Wg[g][(size_t)k * GH + n0 + n]);
      }
  }

  // one-time: private f32 hidden-state carry
  float hreg[8];
#pragma unroll
  for (int v = 0; v < 8; ++v) {
    const int b = wid * 16 + v + 8 * half;
    hreg[v] = h_init[(size_t)b * (2 * GH) + gn];
  }
  __syncthreads();

  unsigned short* hbf_cur = hbfA;
  unsigned short* hbf_nxt = hbfB;

  for (int s = 0; s < GS; ++s) {
    // stage bf16 h (64x512) -> LDS: pure b128 copy, 32 chunks per thread
#pragma unroll 8
    for (int i = 0; i < 32; ++i) {
      const int chunk = tid + i * 128;            // 4096 chunks of 8 ushorts
      const int row = chunk >> 6, c8 = chunk & 63;
      const uint4 d = *(const uint4*)(hbf_cur + (size_t)row * GH + c8 * 8);
      *(uint4*)(Hs + row * 520 + c8 * 8) = d;
    }
    __syncthreads();

    v8f aR = (v8f){0,0,0,0,0,0,0,0}, aZ = (v8f){0,0,0,0,0,0,0,0}, aH = (v8f){0,0,0,0,0,0,0,0};
#pragma unroll 4
    for (int kt = 0; kt < 16; ++kt) {
      const v16bf aF = frag_a(Hs, wid * 16, kt * 32, 520);
      const v16bf bR = frag_b(Wl + 0 * 16 * 520, 0, kt * 32, 520);
      const v16bf bZ = frag_b(Wl + 1 * 16 * 520, 0, kt * 32, 520);
      const v16bf bH = frag_b(Wl + 2 * 16 * 520, 0, kt * 32, 520);
      aR = WMMA_BF16(aF, bR, aR);
      aZ = WMMA_BF16(aF, bZ, aZ);
      aH = WMMA_BF16(aF, bH, aH);
    }

    // elementwise GRU update + writes (+ prefetch next step's gx stream)
#pragma unroll
    for (int v = 0; v < 8; ++v) {
      const int b = wid * 16 + v + 8 * half;
      const size_t gi = ((size_t)b * GS + s) * GH + gn;
      if (s + 1 < GS) {
        __builtin_prefetch(gxr + gi + GH, 0, 1);
        __builtin_prefetch(gxz + gi + GH, 0, 1);
        __builtin_prefetch(gxh + gi + GH, 0, 1);
      }
      const float r  = 1.f / (1.f + __expf(-(gxr[gi] + aR[v])));
      const float z  = 1.f / (1.f + __expf(-(gxz[gi] + aZ[v])));
      const float ht = tanhf(gxh[gi] + r * aH[v]);
      const float hnew = (1.f - z) * hreg[v] + z * ht;
      hreg[v] = hnew;
      hbf_nxt[(size_t)b * GH + gn] = bf1(hnew);
      ys[gi] = hnew;
    }

    // grid-wide barrier (phase = step count in sync[1])
    __threadfence();
    __syncthreads();
    if (tid == 0) {
      const unsigned arrived =
          __hip_atomic_fetch_add(&sync[0], 1u, __ATOMIC_ACQ_REL, __HIP_MEMORY_SCOPE_AGENT);
      if (arrived == gridDim.x - 1) {
        __hip_atomic_store(&sync[0], 0u, __ATOMIC_RELAXED, __HIP_MEMORY_SCOPE_AGENT);
        __hip_atomic_fetch_add(&sync[1], 1u, __ATOMIC_ACQ_REL, __HIP_MEMORY_SCOPE_AGENT);
      } else {
        while (__hip_atomic_load(&sync[1], __ATOMIC_ACQUIRE, __HIP_MEMORY_SCOPE_AGENT) <
               (unsigned)(s + 1)) {
          __builtin_amdgcn_s_sleep(2);
        }
      }
    }
    __syncthreads();
    unsigned short* t = hbf_cur; hbf_cur = hbf_nxt; hbf_nxt = t;
  }
}

// ---------------------------------------------------------------- small kernels
__global__ void init_scan(const float* __restrict__ h_in, int layer,
                          unsigned short* __restrict__ hbf, unsigned* __restrict__ sync) {
  const int i = blockIdx.x * blockDim.x + threadIdx.x;
  if (i < GB * GH) {
    const int b = i >> 9, j = i & 511;
    hbf[i] = bf1(h_in[((size_t)b * 2 + layer) * GH + j]);
  }
  if (blockIdx.x == 0 && threadIdx.x < 2) sync[threadIdx.x] = 0u;
}

__global__ void finals_copy(const float* __restrict__ ys0, const float* __restrict__ inp,
                            float* __restrict__ fin) {
  const int i = blockIdx.x * blockDim.x + threadIdx.x;
  if (i < GB * GH) {
    const int b = i >> 9, j = i & 511;
    fin[(size_t)b * (2 * GH) + j]      = ys0[((size_t)b * GS + (GS - 1)) * GH + j];
    fin[(size_t)b * (2 * GH) + GH + j] = inp[((size_t)b * GS + (GS - 1)) * GH + j];
  }
}

// ---------------------------------------------------------------- launcher
extern "C" void kernel_launch(void* const* d_in, const int* in_sizes, int n_in,
                              void* d_out, int out_size, void* d_ws, size_t ws_size,
                              hipStream_t stream) {
  (void)in_sizes; (void)n_in; (void)out_size; (void)ws_size;
  const float* x  = (const float*)d_in[0];
  const float* h  = (const float*)d_in[1];
  const float* Wr = (const float*)d_in[2];
  const float* br = (const float*)d_in[3];
  const float* Wz = (const float*)d_in[4];
  const float* bz = (const float*)d_in[5];
  const float* Wh = (const float*)d_in[6];
  const float* bh = (const float*)d_in[7];

  const size_t PLANE = (size_t)GB * GS * GH;   // 16,777,216 floats
  float* ws   = (float*)d_ws;
  float* gxr  = ws;
  float* gxz  = ws + PLANE;
  float* gxh  = ws + 2 * PLANE;
  float* ys0  = ws + 3 * PLANE;
  unsigned short* hbf0 = (unsigned short*)(ws + 4 * PLANE);      // 64 KB
  unsigned short* hbf1 = hbf0 + (size_t)GB * GH;                 // 64 KB
  unsigned* sync = (unsigned*)(hbf1 + (size_t)GB * GH);

  float* inp_out = (float*)d_out;                 // (B, S, H)
  float* fin_out = inp_out + PLANE;               // (B, L, H)

  const size_t LW = (size_t)(GD + GH) * GH;       // per-layer weight stride
  const size_t RW = (size_t)GD * GH;              // offset to recurrent rows

  const dim3 ggrid(GH / 128, (GB * GS) / 128, 3); // (4, 256, 3)
  const dim3 sgrid(GH / 16);                      // 32 persistent blocks
  const size_t scanLds = (size_t)(3 * 16 * 520 + GB * 520) * sizeof(unsigned short); // 116480 B

  // ---- layer 0
  gemm_gates<<<ggrid, 256, 0, stream>>>(x, Wr, Wz, Wh, br, bz, bh, gxr, gxz, gxh);
  init_scan<<<(GB * GH + 255) / 256, 256, 0, stream>>>(h, 0, hbf0, sync);
  gru_scan<<<sgrid, 128, scanLds, stream>>>(gxr, gxz, gxh,
                                            Wr + RW, Wz + RW, Wh + RW,
                                            h /* layer 0 */, hbf0, hbf1, ys0, sync);
  // ---- layer 1
  gemm_gates<<<ggrid, 256, 0, stream>>>(ys0, Wr + LW, Wz + LW, Wh + LW,
                                        br + GH, bz + GH, bh + GH, gxr, gxz, gxh);
  init_scan<<<(GB * GH + 255) / 256, 256, 0, stream>>>(h, 1, hbf0, sync);
  gru_scan<<<sgrid, 128, scanLds, stream>>>(gxr, gxz, gxh,
                                            Wr + LW + RW, Wz + LW + RW, Wh + LW + RW,
                                            h + GH /* layer 1 */, hbf0, hbf1, inp_out, sync);
  // ---- final hidden states
  finals_copy<<<(GB * GH + 255) / 256, 256, 0, stream>>>(ys0, inp_out, fin_out);
}
